// LinearAttention_25031069401566
// MI455X (gfx1250) — matrix operand order repaired and checked by reading, not compiled, not run on hardware
//
#include <hip/hip_runtime.h>
#include <hip/hip_bf16.h>
#include <math.h>

// Problem constants (from reference): B=4, S=4096, H=2048, NH=16, HD=128
#define BB  4
#define SS  4096
#define HHH 2048
#define NHH 16
#define HDD 128

typedef __attribute__((ext_vector_type(16))) __bf16 v16bf;
typedef __attribute__((ext_vector_type(8)))  float  v8f;

union Frag16 { v16bf v; uint4 q[2]; };

static __device__ __forceinline__ v8f wmma_bf16(v16bf a, v16bf b, v8f c) {
    // D(16x16,f32) = A(16x32,bf16) * B(32x16,bf16) + C
    return __builtin_amdgcn_wmma_f32_16x16x32_bf16(
        /*neg_a=*/false, a, /*neg_b=*/false, b,
        /*c_mod=*/(short)0, c, /*reuse_a=*/false, /*reuse_b=*/false);
}

// A fragment: 16x32 bf16 tile stored ROW-major [m][k] in LDS (ld bf16 elems).
// Lane layout (ISA 7.12.2): m = lane&15; K groups 0-7/16-23 (+8 for lane>=16).
// Two contiguous 16B reads -> ds_load_b128 x2, no VALU packing.
static __device__ __forceinline__ v16bf load_a_frag(const __bf16* tile, int ld, int lane) {
    const __bf16* p = tile + (size_t)(lane & 15) * ld + ((lane >> 4) << 3);
    Frag16 f;
    f.q[0] = *(const uint4*)(p);        // K = kh .. kh+7
    f.q[1] = *(const uint4*)(p + 16);   // K = 16+kh .. 16+kh+7
    return f.v;
}

// B fragment: 32x16 bf16 tile stored COLUMN-major [n][k] in LDS (ld bf16 elems).
// Lane layout: n = lane&15; K groups 0-7/16-23 (+8 for lane>=16), same as A.
static __device__ __forceinline__ v16bf load_b_frag(const __bf16* tileT, int ld, int lane) {
    const __bf16* p = tileT + (size_t)(lane & 15) * ld + ((lane >> 4) << 3);
    Frag16 f;
    f.q[0] = *(const uint4*)(p);
    f.q[1] = *(const uint4*)(p + 16);
    return f.v;
}

// ---------------------------------------------------------------------------
// fp32 -> bf16 conversion (vectorized, grid-stride)
// ---------------------------------------------------------------------------
__global__ __launch_bounds__(256)
void cvt_f32_bf16(const float* __restrict__ in, __bf16* __restrict__ out, size_t n4) {
    size_t i      = (size_t)blockIdx.x * blockDim.x + threadIdx.x;
    size_t stride = (size_t)gridDim.x * blockDim.x;
    for (; i < n4; i += stride) {
        float4 f = ((const float4*)in)[i];
        __bf16* o = out + i * 4;
        o[0] = (__bf16)f.x; o[1] = (__bf16)f.y;
        o[2] = (__bf16)f.z; o[3] = (__bf16)f.w;
    }
}

// ---------------------------------------------------------------------------
// Fused fp32 -> bf16 + transpose: out[c][r] = (bf16) in[r][c]
// in: R x C fp32 row-major; out: C x R bf16 row-major. 32x32 LDS tile.
// ---------------------------------------------------------------------------
__global__ __launch_bounds__(256)
void cvt_transpose(const float* __restrict__ in, __bf16* __restrict__ out, int R, int C) {
    __shared__ float t[32][33];
    const int bx = blockIdx.x * 32;   // column base
    const int by = blockIdx.y * 32;   // row base
    const int lx = threadIdx.x & 31;
    const int ly = threadIdx.x >> 5;  // 0..7
#pragma unroll
    for (int i = 0; i < 4; ++i) {
        int r = ly + i * 8;
        t[r][lx] = in[(size_t)(by + r) * C + bx + lx];
    }
    __syncthreads();
#pragma unroll
    for (int i = 0; i < 4; ++i) {
        int c = ly + i * 8;
        out[(size_t)(bx + c) * R + by + lx] = (__bf16)t[lx][c];
    }
}

// ---------------------------------------------------------------------------
// C[M,N] = X[M,K] @ W[K,N] + bias, with WT = W^T ([N][K] row-major, ld=K).
// Block tile 128x128, BK=32, 256 threads = 8 waves (2 x 4), wave tile 64x32.
// Double-buffered global->reg->LDS staging + prefetch of the k+2 tiles.
// TRANS_OUT: store C^T (C[col*M + row]) instead of C[row*N + col].
// ---------------------------------------------------------------------------
template <bool SIGMOID, bool TRANS_OUT, typename OutT>
__global__ __launch_bounds__(256)
void gemm_xw(const __bf16* __restrict__ X, const __bf16* __restrict__ WT,
             const float* __restrict__ bias, OutT* __restrict__ C,
             int M, int N, int K) {
    constexpr int BM = 128, BN = 128, BK = 32;
    constexpr int LDA = BK + 8;   // 40 bf16 = 80B rows: 16B aligned, bank-spread
    constexpr int LDB = BK + 8;
    __shared__ __align__(16) __bf16 Xs [BM * LDA];  // [m][k] row-major
    __shared__ __align__(16) __bf16 WsT[BN * LDB];  // [n][k] (W^T tile, row-major)

    const int tid  = threadIdx.x;
    const int lane = tid & 31;
    const int wave = tid >> 5;
    const int wm   = (wave & 1) * 64;
    const int wn   = (wave >> 1) * 32;
    const int bm   = blockIdx.x * BM;
    const int bn   = blockIdx.y * BN;

    uint4 xreg[2], wreg[2];
    auto gfetch = [&](int k0) {
#pragma unroll
        for (int it = 0; it < 2; ++it) {
            const int idx = tid + it * 256;           // 0..511
            const int r   = idx >> 2;                 // 0..127
            const int c8  = (idx & 3) << 3;           // 0,8,16,24 (k)
            xreg[it] = *(const uint4*)(X  + (size_t)(bm + r) * K + k0 + c8);
            wreg[it] = *(const uint4*)(WT + (size_t)(bn + r) * K + k0 + c8);
        }
    };

    gfetch(0);
    v8f acc[4][2] = {};

    for (int k0 = 0; k0 < K; k0 += BK) {
        // Commit staged registers to LDS (all 128-bit stores)
#pragma unroll
        for (int it = 0; it < 2; ++it) {
            const int idx = tid + it * 256;
            const int r   = idx >> 2;
            const int c8  = (idx & 3) << 3;
            *(uint4*)(&Xs [r * LDA + c8]) = xreg[it];
            *(uint4*)(&WsT[r * LDB + c8]) = wreg[it];
        }
        __syncthreads();

        if (k0 + BK < K) gfetch(k0 + BK);             // overlap with WMMAs
        if (k0 + 2 * BK < K) {                        // global_prefetch_b8
            __builtin_prefetch(X  + (size_t)(bm + (tid >> 1)) * K + k0 + 2 * BK, 0, 1);
            __builtin_prefetch(WT + (size_t)(bn + (tid >> 1)) * K + k0 + 2 * BK, 0, 1);
        }

        v16bf bfr[2];
#pragma unroll
        for (int tn = 0; tn < 2; ++tn)
            bfr[tn] = load_b_frag(&WsT[(wn + tn * 16) * LDB], LDB, lane);
#pragma unroll
        for (int tm = 0; tm < 4; ++tm) {
            v16bf afr = load_a_frag(&Xs[(wm + tm * 16) * LDA], LDA, lane);
#pragma unroll
            for (int tn = 0; tn < 2; ++tn)
                acc[tm][tn] = wmma_bf16(afr, bfr[tn], acc[tm][tn]);
        }
        __syncthreads();
    }

    // Epilogue: C/D layout -> lane n = lane&15, rows r + 8*(lane>>4)
    const int n  = lane & 15;
    const int mb = (lane >> 4) << 3;
#pragma unroll
    for (int tm = 0; tm < 4; ++tm) {
#pragma unroll
        for (int tn = 0; tn < 2; ++tn) {
            const int   col = bn + wn + tn * 16 + n;
            const float bv  = bias[col];
#pragma unroll
            for (int r = 0; r < 8; ++r) {
                const int row = bm + wm + tm * 16 + mb + r;
                float v = acc[tm][tn][r] + bv;
                if (SIGMOID) v = 1.0f / (1.0f + __expf(-v));
                if (TRANS_OUT) C[(size_t)col * M + row] = (OutT)v;
                else           C[(size_t)row * N + col] = (OutT)v;
            }
        }
    }
}

// ---------------------------------------------------------------------------
// KV^T per (b,h): consumes kT [H][B*S] and vT [H][B*S] (outputs of the
// TRANS_OUT projections), accumulates KV[d][e] over S, stores KV^T[e][d] bf16.
// All LDS traffic is 128-bit; A/B fragments contiguous.
// ---------------------------------------------------------------------------
__global__ __launch_bounds__(256)
void kv_outer(const __bf16* __restrict__ kT, const __bf16* __restrict__ vT,
              __bf16* __restrict__ kvT) {
    constexpr int LDT = 32 + 8;  // 40
    __shared__ __align__(16) __bf16 KsT[HDD * LDT];  // [d][s]
    __shared__ __align__(16) __bf16 VsT[HDD * LDT];  // [e][s]

    const int bh = blockIdx.x;
    const int b  = bh >> 4;
    const int h  = bh & 15;
    const size_t M = (size_t)BB * SS;
    const __bf16* Kp = kT + (size_t)h * HDD * M + (size_t)b * SS;  // row d, col s
    const __bf16* Vp = vT + (size_t)h * HDD * M + (size_t)b * SS;

    const int tid  = threadIdx.x;
    const int lane = tid & 31;
    const int wave = tid >> 5;
    const int wm   = (wave & 1) * 64;   // d-range
    const int wn   = (wave >> 1) * 32;  // e-range

    uint4 kreg[2], vreg[2];
    auto gfetch = [&](int s0) {
#pragma unroll
        for (int it = 0; it < 2; ++it) {
            const int idx = tid + it * 256;
            const int r   = idx >> 2;        // d or e row, 0..127
            const int c8  = (idx & 3) << 3;  // s offset
            kreg[it] = *(const uint4*)(Kp + (size_t)r * M + s0 + c8);
            vreg[it] = *(const uint4*)(Vp + (size_t)r * M + s0 + c8);
        }
    };

    gfetch(0);
    v8f acc[4][2] = {};

    for (int s0 = 0; s0 < SS; s0 += 32) {
#pragma unroll
        for (int it = 0; it < 2; ++it) {
            const int idx = tid + it * 256;
            const int r   = idx >> 2;
            const int c8  = (idx & 3) << 3;
            *(uint4*)(&KsT[r * LDT + c8]) = kreg[it];
            *(uint4*)(&VsT[r * LDT + c8]) = vreg[it];
        }
        __syncthreads();

        if (s0 + 32 < SS) gfetch(s0 + 32);
        if (s0 + 64 < SS) {
            __builtin_prefetch(Kp + (size_t)(tid >> 1) * M + s0 + 64, 0, 1);
            __builtin_prefetch(Vp + (size_t)(tid >> 1) * M + s0 + 64, 0, 1);
        }

        v16bf bfr[2];
#pragma unroll
        for (int tn = 0; tn < 2; ++tn)
            bfr[tn] = load_b_frag(&VsT[(wn + tn * 16) * LDT], LDT, lane);
#pragma unroll
        for (int tm = 0; tm < 4; ++tm) {
            v16bf afr = load_a_frag(&KsT[(wm + tm * 16) * LDT], LDT, lane);
#pragma unroll
            for (int tn = 0; tn < 2; ++tn)
                acc[tm][tn] = wmma_bf16(afr, bfr[tn], acc[tm][tn]);
        }
        __syncthreads();
    }

    // store KV^T: out[e][d]
    const int n  = lane & 15;
    const int mb = (lane >> 4) << 3;
    __bf16* out = kvT + (size_t)bh * HDD * HDD;
#pragma unroll
    for (int tm = 0; tm < 4; ++tm)
#pragma unroll
        for (int tn = 0; tn < 2; ++tn)
#pragma unroll
            for (int r = 0; r < 8; ++r) {
                const int d = wm + tm * 16 + mb + r;
                const int e = wn + tn * 16 + n;
                out[(size_t)e * HDD + d] = (__bf16)acc[tm][tn][r];
            }
}

// ---------------------------------------------------------------------------
// out[s,e] = (Q[s,:] @ KV) * g ; per block: 128 seq rows of one (b,h).
// kvT is KV^T [e][d] -> exactly the column-major B layout we need.
// ---------------------------------------------------------------------------
__global__ __launch_bounds__(256)
void q_kv_gate(const __bf16* __restrict__ Qb, const __bf16* __restrict__ kvT,
               const __bf16* __restrict__ Gb, __bf16* __restrict__ Ab) {
    constexpr int LDQ = 40, LDKV = 40;
    __shared__ __align__(16) __bf16 Qs  [128 * LDQ];   // [s][d]
    __shared__ __align__(16) __bf16 KVsT[128 * LDKV];  // [e][d]

    const int bh    = blockIdx.y;
    const int b     = bh >> 4;
    const int h     = bh & 15;
    const int sbase = blockIdx.x * 128;
    const __bf16* Qp  = Qb  + (size_t)b * SS * HHH + h * HDD;
    const __bf16* KVp = kvT + (size_t)bh * HDD * HDD;

    const int tid  = threadIdx.x;
    const int lane = tid & 31;
    const int wave = tid >> 5;
    const int wm   = (wave & 1) * 64;
    const int wn   = (wave >> 1) * 32;

    uint4 qreg[2], kvreg[2];
    auto gfetch = [&](int k0) {
#pragma unroll
        for (int it = 0; it < 2; ++it) {
            const int idx = tid + it * 256;
            const int r   = idx >> 2;        // s row / e row
            const int c8  = (idx & 3) << 3;  // d offset
            qreg[it]  = *(const uint4*)(Qp  + (size_t)(sbase + r) * HHH + k0 + c8);
            kvreg[it] = *(const uint4*)(KVp + (size_t)r * HDD + k0 + c8);
        }
    };

    gfetch(0);
    v8f acc[4][2] = {};

    for (int k0 = 0; k0 < HDD; k0 += 32) {
#pragma unroll
        for (int it = 0; it < 2; ++it) {
            const int idx = tid + it * 256;
            const int r   = idx >> 2;
            const int c8  = (idx & 3) << 3;
            *(uint4*)(&Qs  [r * LDQ  + c8]) = qreg[it];
            *(uint4*)(&KVsT[r * LDKV + c8]) = kvreg[it];
        }
        __syncthreads();

        if (k0 + 32 < HDD) gfetch(k0 + 32);

        v16bf bfr[2];
#pragma unroll
        for (int tn = 0; tn < 2; ++tn)
            bfr[tn] = load_b_frag(&KVsT[(wn + tn * 16) * LDKV], LDKV, lane);
#pragma unroll
        for (int tm = 0; tm < 4; ++tm) {
            v16bf afr = load_a_frag(&Qs[(wm + tm * 16) * LDQ], LDQ, lane);
#pragma unroll
            for (int tn = 0; tn < 2; ++tn)
                acc[tm][tn] = wmma_bf16(afr, bfr[tn], acc[tm][tn]);
        }
        __syncthreads();
    }

    const int n  = lane & 15;
    const int mb = (lane >> 4) << 3;
#pragma unroll
    for (int tm = 0; tm < 4; ++tm)
#pragma unroll
        for (int tn = 0; tn < 2; ++tn)
#pragma unroll
            for (int r = 0; r < 8; ++r) {
                const int s = sbase + wm + tm * 16 + mb + r;
                const int e = wn + tn * 16 + n;
                const size_t gi = ((size_t)b * SS + s) * HHH + h * HDD + e;
                const float gv = (float)Gb[gi];
                Ab[gi] = (__bf16)(acc[tm][tn][r] * gv);
            }
}

// ---------------------------------------------------------------------------
// Host-side launch
// ---------------------------------------------------------------------------
extern "C" void kernel_launch(void* const* d_in, const int* in_sizes, int n_in,
                              void* d_out, int out_size, void* d_ws, size_t ws_size,
                              hipStream_t stream) {
    (void)in_sizes; (void)n_in; (void)out_size; (void)ws_size;

    const float* x  = (const float*)d_in[0];
    const float* Wq = (const float*)d_in[1];  const float* bq = (const float*)d_in[2];
    const float* Wk = (const float*)d_in[3];  const float* bk = (const float*)d_in[4];
    const float* Wv = (const float*)d_in[5];  const float* bv = (const float*)d_in[6];
    const float* Wg = (const float*)d_in[7];  const float* bg = (const float*)d_in[8];
    const float* Wo = (const float*)d_in[9];  const float* bo = (const float*)d_in[10];
    float* out = (float*)d_out;

    const size_t M  = (size_t)BB * SS;        // 16384
    const size_t HW = (size_t)HHH * HHH;

    char* p = (char*)d_ws;
    auto carve = [&](size_t elems) -> __bf16* {
        __bf16* q = (__bf16*)p;
        p += (elems * sizeof(__bf16) + 255) & ~(size_t)255;
        return q;
    };
    __bf16* xb   = carve(M * HHH);   // x, bf16 [M][H]
    __bf16* WqT  = carve(HW);        // W^T bf16 [H][H]
    __bf16* WkT  = carve(HW);
    __bf16* WvT  = carve(HW);
    __bf16* WgT  = carve(HW);
    __bf16* WoT  = carve(HW);
    __bf16* qb   = carve(M * HHH);   // q  [M][H]
    __bf16* kTb  = carve(M * HHH);   // k^T [H][M]
    __bf16* vTb  = carve(M * HHH);   // v^T [H][M]
    __bf16* gb   = carve(M * HHH);   // g  [M][H]
    __bf16* kvTb = carve((size_t)BB * NHH * HDD * HDD);  // KV^T per (b,h)
    __bf16* ab   = carve(M * HHH);   // gated attn out [M][H]

    // 1) fp32 -> bf16 (x), fp32 -> bf16 + transpose (weights)
    {
        size_t n4 = (M * HHH) / 4;
        cvt_f32_bf16<<<(unsigned)((n4 + 255) / 256), 256, 0, stream>>>(x, xb, n4);
    }
    dim3 tg(HHH / 32, HHH / 32);
    cvt_transpose<<<tg, 256, 0, stream>>>(Wq, WqT, HHH, HHH);
    cvt_transpose<<<tg, 256, 0, stream>>>(Wk, WkT, HHH, HHH);
    cvt_transpose<<<tg, 256, 0, stream>>>(Wv, WvT, HHH, HHH);
    cvt_transpose<<<tg, 256, 0, stream>>>(Wg, WgT, HHH, HHH);
    cvt_transpose<<<tg, 256, 0, stream>>>(Wo, WoT, HHH, HHH);

    // 2) Projections (q,g normal; k,v transposed outputs for the KV pass)
    dim3 gg((unsigned)(M / 128), (unsigned)(HHH / 128));
    gemm_xw<false, false, __bf16><<<gg, 256, 0, stream>>>(xb, WqT, bq, qb,  (int)M, HHH, HHH);
    gemm_xw<false, true,  __bf16><<<gg, 256, 0, stream>>>(xb, WkT, bk, kTb, (int)M, HHH, HHH);
    gemm_xw<false, true,  __bf16><<<gg, 256, 0, stream>>>(xb, WvT, bv, vTb, (int)M, HHH, HHH);
    gemm_xw<true,  false, __bf16><<<gg, 256, 0, stream>>>(xb, WgT, bg, gb,  (int)M, HHH, HHH);

    // 3) KV^T = (K^T V)^T per (b,h)
    kv_outer<<<BB * NHH, 256, 0, stream>>>(kTb, vTb, kvTb);

    // 4) attn = (Q @ KV) * g
    q_kv_gate<<<dim3(SS / 128, BB * NHH), 256, 0, stream>>>(qb, kvTb, gb, ab);

    // 5) final projection -> fp32 output
    gemm_xw<false, false, float><<<gg, 256, 0, stream>>>(ab, WoT, bo, out, (int)M, HHH, HHH);
}
